// Net_89610197664373
// MI455X (gfx1250) — matrix operand order repaired and checked by reading, not compiled, not run on hardware
//
#include <hip/hip_runtime.h>
#include <hip/hip_bf16.h>
#include <stdint.h>

typedef __attribute__((ext_vector_type(16))) _Float16 v16h;
typedef __attribute__((ext_vector_type(8)))  _Float16 v8h;
typedef __attribute__((ext_vector_type(8)))  float    v8f;

#define TPB 256

// ---------- helpers: order-preserving float<->uint for atomic max ----------
__device__ __forceinline__ unsigned f2ord(float f) {
  unsigned u = __float_as_uint(f);
  return (u & 0x80000000u) ? ~u : (u | 0x80000000u);
}
__device__ __forceinline__ float ord2f(unsigned u) {
  unsigned b = (u & 0x80000000u) ? (u ^ 0x80000000u) : ~u;
  return __uint_as_float(b);
}

// ---------- fills ----------
__global__ void k_fill_u32(unsigned* p, unsigned v, size_t n) {
  size_t i = (size_t)blockIdx.x * blockDim.x + threadIdx.x;
  if (i < n) p[i] = v;
}
__global__ void k_fill_f32(float* p, float v, size_t n) {
  size_t i = (size_t)blockIdx.x * blockDim.x + threadIdx.x;
  if (i < n) p[i] = v;
}

// ---------- stage-1 tiny MLP: out = leaky(z[rows,8] @ w1[8,H], 0.2), f16, K-padded ----------
__global__ void k_stage1(const float* __restrict__ z, const float* __restrict__ w1,
                         _Float16* __restrict__ out, int H, int HIDW, size_t rows) {
  size_t idx = (size_t)blockIdx.x * blockDim.x + threadIdx.x;
  if (idx >= rows * (size_t)HIDW) return;
  int h = (int)(idx % HIDW);
  size_t r = idx / HIDW;
  float v = 0.0f;
  if (h < H) {
    const float* zp = z + r * 8;
#pragma unroll
    for (int c = 0; c < 8; ++c) v += zp[c] * w1[c * H + h];
    v = (v >= 0.0f) ? v : 0.2f * v;
  }
  out[idx] = (_Float16)v;
}

// ---------- pack f32 weight [Ksrc,Nsrc] row-major into WMMA B fragments ----------
// dst[((kt*ntiles+nt)*32 + lane)*16 + i] = B[k, n], n = nt*16 + (lane&15),
// k = kt*32 + (lane>>4)*8 + (i&7) + (i>>3)*16   (zeros outside Ksrc/Nsrc)
__global__ void k_pack_b(const float* __restrict__ src, _Float16* __restrict__ dst,
                         int Ksrc, int ktiles, int Nsrc, int ntiles) {
  int idx = blockIdx.x * blockDim.x + threadIdx.x;
  int total = ktiles * ntiles * 512;
  if (idx >= total) return;
  int i = idx & 15;
  int lane = (idx >> 4) & 31;
  int t = idx >> 9;
  int nt = t % ntiles, kt = t / ntiles;
  int n = nt * 16 + (lane & 15);
  int k = kt * 32 + (lane >> 4) * 8 + (i & 7) + (i >> 3) * 16;
  float v = (k < Ksrc && n < Nsrc) ? src[k * Nsrc + n] : 0.0f;
  dst[idx] = (_Float16)v;
}

// stacked variant: rows [0,K1) from s1, [K1,K1+K2) from s2
__global__ void k_pack_b2(const float* __restrict__ s1, const float* __restrict__ s2,
                          _Float16* __restrict__ dst, int K1, int K2,
                          int ktiles, int Nsrc, int ntiles) {
  int idx = blockIdx.x * blockDim.x + threadIdx.x;
  int total = ktiles * ntiles * 512;
  if (idx >= total) return;
  int i = idx & 15;
  int lane = (idx >> 4) & 31;
  int t = idx >> 9;
  int nt = t % ntiles, kt = t / ntiles;
  int n = nt * 16 + (lane & 15);
  int k = kt * 32 + (lane >> 4) * 8 + (i & 7) + (i >> 3) * 16;
  float v = 0.0f;
  if (n < Nsrc) {
    if (k < K1) v = s1[k * Nsrc + n];
    else if (k < K1 + K2) v = s2[(k - K1) * Nsrc + n];
  }
  dst[idx] = (_Float16)v;
}

// ---------- generic WMMA GEMM: C = act(A[M,Kpad]f16 @ Bpk + bias + exScale*extra) ----------
// one 16x16 output tile per wave; 8 waves per block; grid = (ceil(Mtiles/8), ntiles)
// Two independent accumulator chains (even/odd k-tiles) so consecutive WMMAs have
// no D->C dependency and the compiler can interleave them instead of emitting NOPs.
__device__ __forceinline__ v16h load_afrag(const _Float16* ap) {
  v8h lo = *(const v8h*)ap;          // K = base + [0..7]
  v8h hi = *(const v8h*)(ap + 16);   // K = base + 16 + [0..7]
  v16h a;
#pragma unroll
  for (int i = 0; i < 8; ++i) { a[i] = lo[i]; a[i + 8] = hi[i]; }
  return a;
}

__global__ void k_gemm(const _Float16* __restrict__ A, int lda, int Mtiles,
                       const _Float16* __restrict__ Bpk, int ktiles, int ntiles,
                       const float* __restrict__ bias,
                       const float* __restrict__ extra, float exScale,
                       float negSlope, void* Cout, int outHalf, int ldc) {
  int wave = threadIdx.x >> 5;
  int lane = threadIdx.x & 31;
  int mtile = blockIdx.x * 8 + wave;
  if (mtile >= Mtiles) return;
  int nt = blockIdx.y;
  int hl = lane >> 4;
  size_t mrowA = (size_t)mtile * 16 + (lane & 15);
  const _Float16* arow = A + mrowA * lda + hl * 8;
  const _Float16* brow = Bpk + ((size_t)nt * 32 + lane) * 16;
  size_t bstride = (size_t)ntiles * 512;

  v8f acc0 = {}, acc1 = {};
  int kt = 0;
  for (; kt + 1 < ktiles; kt += 2) {
    v16h a0 = load_afrag(arow + (size_t)kt * 32);
    v16h b0 = *(const v16h*)(brow + (size_t)kt * bstride);
    v16h a1 = load_afrag(arow + (size_t)(kt + 1) * 32);
    v16h b1 = *(const v16h*)(brow + (size_t)(kt + 1) * bstride);
    acc0 = __builtin_amdgcn_wmma_f32_16x16x32_f16(false, a0, false, b0,
                                                  (short)0, acc0, false, false);
    acc1 = __builtin_amdgcn_wmma_f32_16x16x32_f16(false, a1, false, b1,
                                                  (short)0, acc1, false, false);
  }
  if (kt < ktiles) {
    v16h a0 = load_afrag(arow + (size_t)kt * 32);
    v16h b0 = *(const v16h*)(brow + (size_t)kt * bstride);
    acc0 = __builtin_amdgcn_wmma_f32_16x16x32_f16(false, a0, false, b0,
                                                  (short)0, acc0, false, false);
  }

  int n = nt * 16 + (lane & 15);
  int mbase = mtile * 16 + hl * 8;
  float bi = bias ? bias[n] : 0.0f;
#pragma unroll
  for (int r = 0; r < 8; ++r) {
    size_t m = (size_t)(mbase + r);
    float v = acc0[r] + acc1[r] + bi;
    if (extra) v += exScale * extra[m * ldc + n];
    v = (v >= 0.0f) ? v : v * negSlope;
    if (outHalf) ((_Float16*)Cout)[m * ldc + n] = (_Float16)v;
    else         ((float*)Cout)[m * ldc + n] = v;
  }
}

// ---------- segment softmax passes (over edges, C channels) ----------
__global__ void k_segmax(const _Float16* __restrict__ ow, const int* __restrict__ src,
                         const int* __restrict__ dst, unsigned* __restrict__ smax,
                         float* __restrict__ cnt, size_t E, int C) {
  size_t idx = (size_t)blockIdx.x * blockDim.x + threadIdx.x;
  if (idx >= E * (size_t)C) return;
  int c = (int)(idx % C);
  size_t e = idx / C;
  int s = src[e];
  atomicMax(&smax[(size_t)s * C + c], f2ord((float)ow[idx]));
  if (c == 0) atomicAdd(&cnt[dst[e]], 1.0f);
}

__global__ void k_expsum(_Float16* __restrict__ ow, const int* __restrict__ src,
                         const unsigned* __restrict__ smax, float* __restrict__ ssum,
                         size_t E, int C) {
  size_t idx = (size_t)blockIdx.x * blockDim.x + threadIdx.x;
  if (idx >= E * (size_t)C) return;
  int c = (int)(idx % C);
  size_t e = idx / C;
  int s = src[e];
  float m = ord2f(smax[(size_t)s * C + c]);
  float ev = __expf((float)ow[idx] - m);
  ow[idx] = (_Float16)ev;
  atomicAdd(&ssum[(size_t)s * C + c], ev);
}

__global__ void k_msg(const _Float16* __restrict__ ow, const int* __restrict__ src,
                      const int* __restrict__ dst, const float* __restrict__ ssum,
                      const float* __restrict__ xin, float* __restrict__ agg,
                      size_t E, int C) {
  size_t idx = (size_t)blockIdx.x * blockDim.x + threadIdx.x;
  if (idx >= E * (size_t)C) return;
  int c = (int)(idx % C);
  size_t e = idx / C;
  int s = src[e], d = dst[e];
  float en = (float)ow[idx] / (ssum[(size_t)s * C + c] + 1e-16f);
  atomicAdd(&agg[(size_t)d * C + c], en * xin[(size_t)s * C + c]);
}

// ---------- mean + build A' = [agg/cnt | xin] in f16 ----------
__global__ void k_aroot(const float* __restrict__ agg, const float* __restrict__ cnt,
                        const float* __restrict__ xin, _Float16* __restrict__ Ar,
                        size_t N, int C) {
  size_t idx = (size_t)blockIdx.x * blockDim.x + threadIdx.x;
  if (idx >= N * (size_t)C) return;
  int c = (int)(idx % C);
  size_t n = idx / C;
  float a = agg[idx] / fmaxf(cnt[n], 1.0f);
  Ar[n * (size_t)(2 * C) + c] = (_Float16)a;
  Ar[n * (size_t)(2 * C) + C + c] = (_Float16)xin[idx];
}

// ---------- log_softmax over C classes ----------
__global__ void k_logsoftmax(const float* __restrict__ in, float* __restrict__ out,
                             size_t N, int C) {
  size_t n = (size_t)blockIdx.x * blockDim.x + threadIdx.x;
  if (n >= N) return;
  const float* p = in + n * C;
  float m = p[0];
  for (int j = 1; j < C; ++j) m = fmaxf(m, p[j]);
  float s = 0.0f;
  for (int j = 0; j < C; ++j) s += __expf(p[j] - m);
  float ls = logf(s);
  for (int j = 0; j < C; ++j) out[n * C + j] = p[j] - m - ls;
}

struct LayerPtrs {
  const float *hm_w1, *hm_w2, *hm_b2, *ham_w1, *ham_w2, *ham_b2;
  const float *lin_l_w, *lin_l_b, *lin_r_w;
};

static inline unsigned gdim(size_t n) { return (unsigned)((n + TPB - 1) / TPB); }

extern "C" void kernel_launch(void* const* d_in, const int* in_sizes, int n_in,
                              void* d_out, int out_size, void* d_ws, size_t ws_size,
                              hipStream_t stream) {
  const int F = 128;  // in_ch for both layers (F_IN = HID = 128)
  const float* x  = (const float*)d_in[0];
  const int*   ei = (const int*)d_in[1];
  const float* kr = (const float*)d_in[2];
  const float* ep = (const float*)d_in[3];
  size_t N = (size_t)in_sizes[0] / F;
  size_t E = (size_t)in_sizes[1] / 2;
  if (n_in < 22) return;

  LayerPtrs P1 = {(const float*)d_in[4],  (const float*)d_in[5],  (const float*)d_in[6],
                  (const float*)d_in[7],  (const float*)d_in[8],  (const float*)d_in[9],
                  (const float*)d_in[10], (const float*)d_in[11], (const float*)d_in[12]};
  LayerPtrs P2 = {(const float*)d_in[13], (const float*)d_in[14], (const float*)d_in[15],
                  (const float*)d_in[16], (const float*)d_in[17], (const float*)d_in[18],
                  (const float*)d_in[19], (const float*)d_in[20], (const float*)d_in[21]};

  // ---- workspace layout ----
  char* base = (char*)d_ws;
  size_t off = 0;
  auto take = [&](size_t bytes) -> char* {
    char* p = base + off;
    off = (off + bytes + 255) & ~(size_t)255;
    return p;
  };
  _Float16* ow    = (_Float16*)take(E * F * 2);        // edge logits -> exp values
  _Float16* eh    = (_Float16*)take(E * F * 2);        // edge hidden (K-padded)
  unsigned* smax  = (unsigned*)take(N * F * 4);
  size_t zoff0 = off;
  float*    ssum  = (float*)take(N * F * 4);
  float*    agg   = (float*)take(N * F * 4);
  float*    cnt   = (float*)take(N * 4);
  size_t zbytes = off - zoff0;
  _Float16* Ar    = (_Float16*)take(N * 2 * F * 2);    // [agg_norm | x] f16
  _Float16* nh    = (_Float16*)take(N * F * 2);        // ham hidden (K-padded)
  float*    pw    = (float*)take(N * F * 4);
  float*    h1    = (float*)take(N * F * 4);
  float*    lgt   = (float*)take(N * 16 * 4);
  _Float16* pkE   = (_Float16*)take(4 * 8 * 512 * 2);  // edge w2 fragments
  _Float16* pkH   = (_Float16*)take(4 * 8 * 512 * 2);  // ham w2 fragments
  _Float16* pkL   = (_Float16*)take(8 * 8 * 512 * 2);  // [lin_l; lin_r] fragments
  if (off > ws_size) return;  // workspace too small; nothing safe to do

  auto run_layer = [&](const LayerPtrs& P, int outc, const float* xin, float* hout,
                       bool relu) {
    int H = outc;                    // hidden width of both small MLPs
    int HIDW = (H + 31) & ~31;       // pad K to multiple of 32
    int ktE = HIDW / 32;
    int ntO = outc / 16;
    unsigned Mt_e = (unsigned)(E / 16), Mt_n = (unsigned)((N + 15) / 16);

    // weight packing
    k_pack_b<<<gdim(ktE * 8 * 512), TPB, 0, stream>>>(P.hm_w2, pkE, H, ktE, F, 8);
    k_pack_b<<<gdim(ktE * ntO * 512), TPB, 0, stream>>>(P.ham_w2, pkH, H, ktE, outc, ntO);
    k_pack_b2<<<gdim(8 * ntO * 512), TPB, 0, stream>>>(P.lin_l_w, P.lin_r_w, pkL,
                                                       F, F, 8, outc, ntO);
    // stage-1 tiny MLPs (K=8), f16 padded outputs
    k_stage1<<<gdim(E * HIDW), TPB, 0, stream>>>(kr, P.hm_w1, eh, H, HIDW, E);
    k_stage1<<<gdim(N * HIDW), TPB, 0, stream>>>(ep, P.ham_w1, nh, H, HIDW, N);

    // edge weight logits: ow = eh @ w2 + b2   [E, F] (f16 out)
    k_gemm<<<dim3((Mt_e + 7) / 8, 8), TPB, 0, stream>>>(
        eh, HIDW, (int)Mt_e, pkE, ktE, 8, P.hm_b2, nullptr, 0.0f, 1.0f, ow, 1, F);

    // segment softmax over src + mean-aggregate at dst
    k_fill_u32<<<gdim(N * F), TPB, 0, stream>>>(smax, 0x007FFFFFu /* ord(-inf) */, N * F);
    k_fill_f32<<<gdim(zbytes / 4), TPB, 0, stream>>>(ssum, 0.0f, zbytes / 4);
    k_segmax<<<gdim(E * F), TPB, 0, stream>>>(ow, ei, ei + E, smax, cnt, E, F);
    k_expsum<<<gdim(E * F), TPB, 0, stream>>>(ow, ei, smax, ssum, E, F);
    k_msg<<<gdim(E * F), TPB, 0, stream>>>(ow, ei, ei + E, ssum, xin, agg, E, F);
    k_aroot<<<gdim(N * F), TPB, 0, stream>>>(agg, cnt, xin, Ar, N, F);

    // node bias: pw = leaky(nh @ ham_w2 + ham_b2, 0.01)   [N, outc]
    k_gemm<<<dim3((Mt_n + 7) / 8, ntO), TPB, 0, stream>>>(
        nh, HIDW, (int)Mt_n, pkH, ktE, ntO, P.ham_b2, nullptr, 0.0f, 0.01f, pw, 0, outc);

    // out = [agg_norm | x] @ [lin_l; lin_r] + lin_l_b + 1.0*pw, (ReLU for layer 1)
    k_gemm<<<dim3((Mt_n + 7) / 8, ntO), TPB, 0, stream>>>(
        Ar, 2 * F, (int)Mt_n, pkL, 8, ntO, P.lin_l_b, pw, 1.0f,
        relu ? 0.0f : 1.0f, hout, 0, outc);
  };

  run_layer(P1, 128, x, h1, true);
  run_layer(P2, 16, h1, lgt, false);
  k_logsoftmax<<<gdim(N), TPB, 0, stream>>>(lgt, (float*)d_out, N, 16);
  (void)out_size;
}
